// TransformerBlockQuantum_65481071398155
// MI455X (gfx1250) — matrix operand order repaired and checked by reading, not compiled, not run on hardware
//
#include <hip/hip_runtime.h>
#include <hip/hip_bf16.h>
#include <math.h>

#define EMBED 768
#define HEADS 12
#define DKH   64
#define FFN   3072
#define NQ    8
#define BATCH 4
#define SEQ   2048
#define TOKENS (BATCH*SEQ)

typedef __bf16 bf16_t;
typedef __attribute__((ext_vector_type(16))) __bf16 v16bf;
typedef __attribute__((ext_vector_type(8)))  float  v8f;
typedef unsigned int u32x4 __attribute__((ext_vector_type(4)));
typedef int          i32x8 __attribute__((ext_vector_type(8)));
typedef int          i32x4 __attribute__((ext_vector_type(4)));

union FragBF { v16bf v; uint4 q[2]; };

// A-matrix 16x32 bf16 fragment: lane&15 = row M; lane-half selects K chunks
// {0..7,16..23} (lanes 0-15) / {8..15,24..31} (lanes 16-31).  [CDNA5 ISA]
__device__ __forceinline__ v16bf load_a_frag(const bf16_t* base, int ld, int m0, int k0, int lane) {
  const bf16_t* p = base + (size_t)(m0 + (lane & 15)) * ld + (k0 + ((lane >> 4) << 3));
  FragBF f;
  f.q[0] = *(const uint4*)(p);
  f.q[1] = *(const uint4*)(p + 16);
  return f.v;
}

// B-matrix 32x16 bf16 fragment: lane&15 = col N; lane-half selects K chunks
// {0..7,16..23} (lanes 0-15) / {8..15,24..31} (lanes 16-31).  Rows of base are the N dimension.
__device__ __forceinline__ v16bf load_b_frag(const bf16_t* base, int ld, int n0, int k0, int lane) {
  const bf16_t* p = base + (size_t)(n0 + (lane & 15)) * ld + (k0 + ((lane >> 4) << 3));
  FragBF f;
  f.q[0] = *(const uint4*)(p);
  f.q[1] = *(const uint4*)(p + 16);
  return f.v;
}

__device__ __forceinline__ v8f wmma_bf16(v16bf a, v16bf b, v8f c) {
  return __builtin_amdgcn_wmma_f32_16x16x32_bf16(false, a, false, b, (short)0, c, false, false);
}

// ---- DPP16 butterfly reductions within each 16-lane half (VALU, no LDS) ----
template <int CTRL>
__device__ __forceinline__ float dppf(float v) {
  return __builtin_bit_cast(float,
      __builtin_amdgcn_update_dpp(0, __builtin_bit_cast(int, v), CTRL, 0xf, 0xf, true));
}
__device__ __forceinline__ float rowmax16(float v) {
  v = fmaxf(v, dppf<0xB1>(v));    // quad_perm [1,0,3,2]  (xor 1)
  v = fmaxf(v, dppf<0x4E>(v));    // quad_perm [2,3,0,1]  (xor 2)
  v = fmaxf(v, dppf<0x141>(v));   // row_half_mirror      (xor 7 within 8)
  v = fmaxf(v, dppf<0x140>(v));   // row_mirror           (xor 15 within 16)
  return v;
}
__device__ __forceinline__ float rowsum16(float v) {
  v += dppf<0xB1>(v);
  v += dppf<0x4E>(v);
  v += dppf<0x141>(v);
  v += dppf<0x140>(v);
  return v;
}

// ---- Tensor Data Mover: async 2D tile load Global -> LDS  [CDNA5 ISA]
__device__ __forceinline__ unsigned lds_bytes(const void* p) {
  // §10.2: LDS aperture flat address low 32 bits are the LDS byte offset.
  return (unsigned)(unsigned long long)p;
}
__device__ __forceinline__ void tdm_load_2d(unsigned lds_addr, const bf16_t* gaddr,
                                            int tensor_d0, int tensor_d1,
                                            long long stride0,
                                            int tile_d0, int tile_d1) {
  const unsigned long long ga = (unsigned long long)gaddr;
  u32x4 g0;
  g0[0] = 1u;                                              // count=1, user descriptor
  g0[1] = lds_addr;                                        // LDS byte address
  g0[2] = (unsigned)(ga & 0xffffffffu);                    // global_addr[31:0]
  g0[3] = (unsigned)((ga >> 32) & 0x1ffffffu) | (2u << 30);// global_addr[56:32] | type=2
  i32x8 g1;
  g1[0] = (1 << 16);                                       // workgroup_mask=0, data_size=2B
  g1[1] = (tensor_d0 & 0xffff) << 16;                      // tensor_dim0[15:0]
  g1[2] = ((unsigned)tensor_d0 >> 16) | ((tensor_d1 & 0xffff) << 16);
  g1[3] = ((unsigned)tensor_d1 >> 16) | (tile_d0 << 16);   // tile_dim0
  g1[4] = (tile_d1 & 0xffff);                              // tile_dim1, tile_dim2=0
  g1[5] = (int)(stride0 & 0xffffffffll);                   // tensor_dim0_stride[31:0]
  g1[6] = (int)((stride0 >> 32) & 0xffffll);               // stride[47:32], dim1_stride lo=0
  g1[7] = 0;
  i32x4 gz;
  gz[0] = 0; gz[1] = 0; gz[2] = 0; gz[3] = 0;
  i32x8 gz8;
  gz8[0] = 0; gz8[1] = 0; gz8[2] = 0; gz8[3] = 0;
  gz8[4] = 0; gz8[5] = 0; gz8[6] = 0; gz8[7] = 0;
  // amdgpu-toolchain (clang-23) 6-arg form: (g0, g1, g2, g3, extra, cpol)
  __builtin_amdgcn_tensor_load_to_lds(g0, g1, gz, gz, gz8, 0);
}

__global__ void f32_to_bf16_k(const float* __restrict__ in, bf16_t* __restrict__ out, size_t n) {
  size_t i = (size_t)blockIdx.x * blockDim.x + threadIdx.x;
  if (i < n) out[i] = (bf16_t)in[i];
}

enum { OUT_F32_RES = 0, OUT_BF16_HEAD = 1, OUT_BF16_VT = 2 };

template <int MODE>
__device__ __forceinline__ void store_tile16(void* outp, const float* __restrict__ residual,
                                             int N, int mt, int nt, int laneN, int hi, v8f acc) {
  const int n = nt + laneN;
#pragma unroll
  for (int r = 0; r < 8; ++r) {
    const int m = mt + r + (hi << 3);       // C layout: vgpr r -> row r / r+8
    if (MODE == OUT_F32_RES) {
      float* O = (float*)outp;
      O[(size_t)m * N + n] = acc[r] + residual[(size_t)m * N + n];
    } else {
      const int bb = m >> 11, s = m & (SEQ - 1);
      const int hh = n >> 6, d = n & (DKH - 1);
      bf16_t* O = (bf16_t*)outp;
      if (MODE == OUT_BF16_HEAD)
        O[(((size_t)(bb * HEADS + hh) * SEQ) + s) * DKH + d] = (bf16_t)acc[r];
      else  // V transposed: [B,H,DKH,SEQ]
        O[(((size_t)(bb * HEADS + hh) * DKH) + d) * SEQ + s] = (bf16_t)acc[r];
    }
  }
}

// C[M,N] = A[M,K] @ W[N,K]^T + bias[N].
// Block: 128(M) x 64(N) tile, 8 waves each computing 32x32.
// TDM double-buffers 128x32 A and 64x32 B tiles into LDS (TENSORcnt pipelined).
template <int MODE>
__global__ __launch_bounds__(256) void gemm_tdm_wmma(
    const bf16_t* __restrict__ A, const bf16_t* __restrict__ W,
    const float* __restrict__ bias, const float* __restrict__ residual,
    void* __restrict__ outp, int M, int N, int K) {
  __shared__ bf16_t ldsA[2][128 * 32];
  __shared__ bf16_t ldsB[2][64 * 32];

  const int lane = threadIdx.x & 31;
  const int wave = threadIdx.x >> 5;
  const int laneN = lane & 15;
  const int hi = lane >> 4;
  const int m0 = blockIdx.y * 128;
  const int n0 = blockIdx.x * 64;
  const int wm = (wave & 3) * 32;
  const int wn = (wave >> 2) * 32;

  const float bn0 = bias[n0 + wn + laneN];
  const float bn1 = bias[n0 + wn + 16 + laneN];
  v8f acc00, acc01, acc10, acc11;
#pragma unroll
  for (int r = 0; r < 8; ++r) { acc00[r] = bn0; acc01[r] = bn1; acc10[r] = bn0; acc11[r] = bn1; }

  const bool issuer = (wave == 0);
  if (issuer) {
    tdm_load_2d(lds_bytes(ldsA[0]), A + (size_t)m0 * K, K, M, K, 32, 128);
    tdm_load_2d(lds_bytes(ldsB[0]), W + (size_t)n0 * K, K, N, K, 32, 64);
  }

  int buf = 0;
  for (int k0 = 0; k0 < K; k0 += 32) {
    if (issuer) {
      if (k0 + 32 < K) {
        tdm_load_2d(lds_bytes(ldsA[buf ^ 1]), A + (size_t)m0 * K + (k0 + 32), K, M, K, 32, 128);
        tdm_load_2d(lds_bytes(ldsB[buf ^ 1]), W + (size_t)n0 * K + (k0 + 32), K, N, K, 32, 64);
        __builtin_amdgcn_s_wait_tensorcnt(2);   // current pair done, next pair in flight
      } else {
        __builtin_amdgcn_s_wait_tensorcnt(0);
      }
    }
    __syncthreads();
    const bf16_t* At = ldsA[buf];
    const bf16_t* Bt = ldsB[buf];
    v16bf a0 = load_a_frag(At, 32, wm, 0, lane);
    v16bf a1 = load_a_frag(At, 32, wm + 16, 0, lane);
    v16bf b0 = load_b_frag(Bt, 32, wn, 0, lane);
    v16bf b1 = load_b_frag(Bt, 32, wn + 16, 0, lane);
    acc00 = wmma_bf16(a0, b0, acc00);
    acc01 = wmma_bf16(a0, b1, acc01);
    acc10 = wmma_bf16(a1, b0, acc10);
    acc11 = wmma_bf16(a1, b1, acc11);
    __syncthreads();     // buffer `buf` free for overwrite next iteration
    buf ^= 1;
  }

  store_tile16<MODE>(outp, residual, N, m0 + wm,      n0 + wn,      laneN, hi, acc00);
  store_tile16<MODE>(outp, residual, N, m0 + wm,      n0 + wn + 16, laneN, hi, acc01);
  store_tile16<MODE>(outp, residual, N, m0 + wm + 16, n0 + wn,      laneN, hi, acc10);
  store_tile16<MODE>(outp, residual, N, m0 + wm + 16, n0 + wn + 16, laneN, hi, acc11);
}

// Flash attention: one wave per 16 query rows; key tiles of 32.
__global__ __launch_bounds__(128) void attn_flash(
    const bf16_t* __restrict__ Q, const bf16_t* __restrict__ Kh,
    const bf16_t* __restrict__ VT, const int* __restrict__ mask,
    bf16_t* __restrict__ ctx) {
  const int lane = threadIdx.x & 31;
  const int wave = threadIdx.x >> 5;
  const int hi = lane >> 4;
  const int laneN = lane & 15;
  const int b = blockIdx.z, h = blockIdx.y;
  const int q0 = (blockIdx.x * 4 + wave) * 16;

  const bf16_t* Qp = Q  + (size_t)(b * HEADS + h) * SEQ * DKH;
  const bf16_t* Kp = Kh + (size_t)(b * HEADS + h) * SEQ * DKH;
  const bf16_t* Vp = VT + (size_t)(b * HEADS + h) * DKH * SEQ;
  const int* mrow = mask + (size_t)b * SEQ;

  __shared__ bf16_t Pbuf[4][16 * 32];
  bf16_t* P = Pbuf[wave];

  v16bf qa0 = load_a_frag(Qp, DKH, q0, 0, lane);
  v16bf qa1 = load_a_frag(Qp, DKH, q0, 32, lane);

  v8f acc0, acc1, acc2, acc3;
  float rmax[8], rsum[8];
#pragma unroll
  for (int r = 0; r < 8; ++r) {
    acc0[r] = 0.f; acc1[r] = 0.f; acc2[r] = 0.f; acc3[r] = 0.f;
    rmax[r] = -INFINITY; rsum[r] = 0.f;
  }

  for (int kt = 0; kt < SEQ; kt += 32) {
    v8f s0, s1;
#pragma unroll
    for (int r = 0; r < 8; ++r) { s0[r] = 0.f; s1[r] = 0.f; }
    s0 = wmma_bf16(qa0, load_b_frag(Kp, DKH, kt,      0,  lane), s0);
    s0 = wmma_bf16(qa1, load_b_frag(Kp, DKH, kt,      32, lane), s0);
    s1 = wmma_bf16(qa0, load_b_frag(Kp, DKH, kt + 16, 0,  lane), s1);
    s1 = wmma_bf16(qa1, load_b_frag(Kp, DKH, kt + 16, 32, lane), s1);

    const float madd0 = (mrow[kt + laneN]      == 0) ? -1e9f : 0.f;
    const float madd1 = (mrow[kt + 16 + laneN] == 0) ? -1e9f : 0.f;

    float p0[8], p1[8];
#pragma unroll
    for (int r = 0; r < 8; ++r) {
      p0[r] = s0[r] * 0.125f + madd0;   // 1/sqrt(64)
      p1[r] = s1[r] * 0.125f + madd1;
      // rows live per (vgpr, lane-half): DPP16 butterfly all-reduce, no LDS.
      const float mx = rowmax16(fmaxf(p0[r], p1[r]));
      const float mnew = fmaxf(rmax[r], mx);
      const float scale = __expf(rmax[r] - mnew);
      const float e0 = __expf(p0[r] - mnew);
      const float e1 = __expf(p1[r] - mnew);
      rsum[r] = rsum[r] * scale + rowsum16(e0 + e1);
      rmax[r] = mnew;
      acc0[r] *= scale; acc1[r] *= scale; acc2[r] *= scale; acc3[r] *= scale;
      p0[r] = e0; p1[r] = e1;
    }

    // C-layout -> A-layout transpose of P through LDS (per-wave region).
#pragma unroll
    for (int r = 0; r < 8; ++r) {
      const int m = r + (hi << 3);
      P[m * 32 + laneN]      = (bf16_t)p0[r];
      P[m * 32 + 16 + laneN] = (bf16_t)p1[r];
    }
    asm volatile("s_wait_dscnt 0x0" ::: "memory");
    FragBF pf;
    const bf16_t* pp = P + laneN * 32 + (hi << 3);
    pf.q[0] = *(const uint4*)(pp);
    pf.q[1] = *(const uint4*)(pp + 16);
    const v16bf pa = pf.v;

    acc0 = wmma_bf16(pa, load_b_frag(Vp, SEQ, 0,  kt, lane), acc0);
    acc1 = wmma_bf16(pa, load_b_frag(Vp, SEQ, 16, kt, lane), acc1);
    acc2 = wmma_bf16(pa, load_b_frag(Vp, SEQ, 32, kt, lane), acc2);
    acc3 = wmma_bf16(pa, load_b_frag(Vp, SEQ, 48, kt, lane), acc3);
  }

#pragma unroll
  for (int r = 0; r < 8; ++r) {
    const int m = q0 + r + (hi << 3);
    const float inv = 1.f / rsum[r];
    bf16_t* crow = ctx + ((size_t)b * SEQ + m) * EMBED + h * DKH;
    crow[0  + laneN] = (bf16_t)(acc0[r] * inv);
    crow[16 + laneN] = (bf16_t)(acc1[r] * inv);
    crow[32 + laneN] = (bf16_t)(acc2[r] * inv);
    crow[48 + laneN] = (bf16_t)(acc3[r] * inv);
  }
}

__global__ __launch_bounds__(256) void layernorm_k(
    const float* __restrict__ y, const float* __restrict__ g,
    const float* __restrict__ be, float* __restrict__ out) {
  const int row = blockIdx.x;
  const float* yr = y + (size_t)row * EMBED;
  float s = 0.f, s2 = 0.f;
  for (int i = threadIdx.x; i < EMBED; i += 256) { const float v = yr[i]; s += v; s2 += v * v; }
#pragma unroll
  for (int off = 1; off < 32; off <<= 1) { s += __shfl_xor(s, off, 32); s2 += __shfl_xor(s2, off, 32); }
  __shared__ float wsum[8], wsum2[8];
  if ((threadIdx.x & 31) == 0) { wsum[threadIdx.x >> 5] = s; wsum2[threadIdx.x >> 5] = s2; }
  __syncthreads();
  s = 0.f; s2 = 0.f;
#pragma unroll
  for (int w = 0; w < 8; ++w) { s += wsum[w]; s2 += wsum2[w]; }
  const float mu = s * (1.f / EMBED);
  const float var = s2 * (1.f / EMBED) - mu * mu;
  const float rstd = rsqrtf(var + 1e-5f);
  for (int i = threadIdx.x; i < EMBED; i += 256)
    out[(size_t)row * EMBED + i] = (yr[i] - mu) * rstd * g[i] + be[i];
}

// h = relu(cos(x1[:, :8] + theta) @ w1^T + b1), written as bf16. Inner dim 8.
__global__ __launch_bounds__(256) void ffn1_k(
    const float* __restrict__ x1, const float* __restrict__ theta,
    const float* __restrict__ w1, const float* __restrict__ b1,
    bf16_t* __restrict__ hout) {
  const int row = blockIdx.x;
  __shared__ float meas[NQ];
  if (threadIdx.x < NQ)
    meas[threadIdx.x] = cosf(x1[(size_t)row * EMBED + threadIdx.x] + theta[threadIdx.x]);
  __syncthreads();
  float m[NQ];
#pragma unroll
  for (int i = 0; i < NQ; ++i) m[i] = meas[i];
  for (int j = threadIdx.x; j < FFN; j += 256) {
    const float4* wr = (const float4*)(w1 + (size_t)j * NQ);
    const float4 wa = wr[0], wb = wr[1];
    float acc = b1[j];
    acc += m[0] * wa.x + m[1] * wa.y + m[2] * wa.z + m[3] * wa.w;
    acc += m[4] * wb.x + m[5] * wb.y + m[6] * wb.z + m[7] * wb.w;
    hout[(size_t)row * FFN + j] = (bf16_t)fmaxf(acc, 0.f);
  }
}

extern "C" void kernel_launch(void* const* d_in, const int* in_sizes, int n_in,
                              void* d_out, int out_size, void* d_ws, size_t ws_size,
                              hipStream_t stream) {
  const float* x     = (const float*)d_in[0];
  const int*   mask  = (const int*)  d_in[1];
  const float* wq    = (const float*)d_in[2];
  const float* bq    = (const float*)d_in[3];
  const float* wk    = (const float*)d_in[4];
  const float* bk    = (const float*)d_in[5];
  const float* wv    = (const float*)d_in[6];
  const float* bv    = (const float*)d_in[7];
  const float* wo    = (const float*)d_in[8];
  const float* bo    = (const float*)d_in[9];
  const float* g1    = (const float*)d_in[10];
  const float* be1   = (const float*)d_in[11];
  const float* g2    = (const float*)d_in[12];
  const float* be2   = (const float*)d_in[13];
  const float* theta = (const float*)d_in[14];
  const float* w1    = (const float*)d_in[15];
  const float* b1    = (const float*)d_in[16];
  const float* w2    = (const float*)d_in[17];
  const float* b2    = (const float*)d_in[18];

  char* cur = (char*)d_ws;
  auto alloc = [&](size_t bytes) -> void* {
    void* p = (void*)cur; cur += (bytes + 255) & ~(size_t)255; return p;
  };
  const size_t tokE2 = (size_t)TOKENS * EMBED * sizeof(bf16_t);
  const size_t tokE4 = (size_t)TOKENS * EMBED * sizeof(float);

  bf16_t* wqb = (bf16_t*)alloc((size_t)EMBED * EMBED * 2);
  bf16_t* wkb = (bf16_t*)alloc((size_t)EMBED * EMBED * 2);
  bf16_t* wvb = (bf16_t*)alloc((size_t)EMBED * EMBED * 2);
  bf16_t* wob = (bf16_t*)alloc((size_t)EMBED * EMBED * 2);
  bf16_t* w2b = (bf16_t*)alloc((size_t)EMBED * FFN * 2);
  float*  y1  = (float*) alloc(tokE4);      // x + attn_out, later reused as y2
  float*  x1  = (float*) alloc(tokE4);      // LN1 output
  bf16_t* xb  = (bf16_t*)alloc(tokE2);      // bf16 x, later reused as ctx
  char*  qkv0 = cur;
  bf16_t* Qh  = (bf16_t*)alloc(tokE2);      // [B,H,S,64]
  bf16_t* Kh  = (bf16_t*)alloc(tokE2);      // [B,H,S,64]
  bf16_t* VTb = (bf16_t*)alloc(tokE2);      // [B,H,64,S]
  bf16_t* ctxb = xb;
  float*  y2   = y1;
  bf16_t* hb   = (bf16_t*)qkv0;             // [TOKENS, FFN], reuses Q/K/VT
  const size_t hb_bytes = (size_t)TOKENS * FFN * 2;
  if ((size_t)(cur - qkv0) < hb_bytes) cur = qkv0 + hb_bytes;

  auto cvt = [&](const float* src, bf16_t* dst, size_t n) {
    f32_to_bf16_k<<<(unsigned)((n + 255) / 256), 256, 0, stream>>>(src, dst, n);
  };
  cvt(x,  xb,  (size_t)TOKENS * EMBED);
  cvt(wq, wqb, (size_t)EMBED * EMBED);
  cvt(wk, wkb, (size_t)EMBED * EMBED);
  cvt(wv, wvb, (size_t)EMBED * EMBED);
  cvt(wo, wob, (size_t)EMBED * EMBED);
  cvt(w2, w2b, (size_t)EMBED * FFN);

  const dim3 ggrid(EMBED / 64, TOKENS / 128);   // (12, 64)

  gemm_tdm_wmma<OUT_BF16_HEAD><<<ggrid, 256, 0, stream>>>(
      xb, wqb, bq, nullptr, Qh, TOKENS, EMBED, EMBED);
  gemm_tdm_wmma<OUT_BF16_HEAD><<<ggrid, 256, 0, stream>>>(
      xb, wkb, bk, nullptr, Kh, TOKENS, EMBED, EMBED);
  gemm_tdm_wmma<OUT_BF16_VT><<<ggrid, 256, 0, stream>>>(
      xb, wvb, bv, nullptr, VTb, TOKENS, EMBED, EMBED);

  attn_flash<<<dim3(SEQ / 64, HEADS, BATCH), 128, 0, stream>>>(Qh, Kh, VTb, mask, ctxb);

  gemm_tdm_wmma<OUT_F32_RES><<<ggrid, 256, 0, stream>>>(
      ctxb, wob, bo, x, y1, TOKENS, EMBED, EMBED);
  layernorm_k<<<TOKENS, 256, 0, stream>>>(y1, g1, be1, x1);

  ffn1_k<<<TOKENS, 256, 0, stream>>>(x1, theta, w1, b1, hb);
  gemm_tdm_wmma<OUT_F32_RES><<<ggrid, 256, 0, stream>>>(
      hb, w2b, b2, x1, y2, TOKENS, EMBED, FFN);
  layernorm_k<<<TOKENS, 256, 0, stream>>>(y2, g2, be2, (float*)d_out);
}
